// PointSpatialConv_10333691314668
// MI455X (gfx1250) — matrix operand order repaired and checked by reading, not compiled, not run on hardware
//
#include <hip/hip_runtime.h>

// Problem constants (from reference setup_inputs)
#define B_   8
#define T_   3
#define N_   4096
#define M_   1024          // N / STRIDE
#define NS_  32            // NSAMPLE
#define C0   64
#define C1   128
#define C2   256
#define R2_  0.25f         // RADIUS^2
#define RPF  ((size_t)B_ * M_ * NS_)   // 262144 rows per frame

typedef _Float16 v16h  __attribute__((ext_vector_type(16)));
typedef _Float16 half8 __attribute__((ext_vector_type(8)));
typedef float    v8f   __attribute__((ext_vector_type(8)));

// Async global->LDS 16B copy (CDNA5 GLOBAL_LOAD_ASYNC_TO_LDS_B128, ASYNCcnt).
// ldsByteOff = low 32 bits of the flat shared-memory pointer (LDS aperture
// truncates to addr[31:0] per ISA 10.2). Per-lane semantics.
__device__ __forceinline__ void async_copy_b128(unsigned ldsByteOff, const void* gsrc)
{
    asm volatile("global_load_async_to_lds_b128 %0, %1, off"
                 :: "v"(ldsByteOff), "v"(gsrc) : "memory");
}
__device__ __forceinline__ void async_wait0()
{
    asm volatile("s_wait_asynccnt 0x0" ::: "memory");
}

// ---------------------------------------------------------------------------
// 1) Farthest point sampling: one block per (t,b). Point cloud lives in LDS.
//    Writes new_xyz to workspace (t-major) and to d_out (B,T,M,3).
// ---------------------------------------------------------------------------
__global__ __launch_bounds__(1024) void fps_kernel(const float* __restrict__ xyzs,
                                                   float* __restrict__ newxyz,
                                                   float* __restrict__ out_newxyz)
{
    const int t = blockIdx.x / B_;
    const int b = blockIdx.x % B_;
    const float* xyz = xyzs + ((size_t)b * T_ + t) * N_ * 3;

    __shared__ float sx[N_], sy[N_], sz[N_];
    __shared__ float redV[32];
    __shared__ int   redI[32];
    __shared__ int   fpsIdx[M_];
    __shared__ int   sel;

    const int tid = threadIdx.x;
    for (int p = tid; p < N_; p += 1024) {
        sx[p] = xyz[p * 3 + 0];
        sy[p] = xyz[p * 3 + 1];
        sz[p] = xyz[p * 3 + 2];
    }
    if (tid == 0) fpsIdx[0] = 0;
    __syncthreads();

    float mind[4];
#pragma unroll
    for (int i = 0; i < 4; ++i) mind[i] = 1e10f;

    int last = 0;
    const int lane = tid & 31, wid = tid >> 5;

    for (int s = 1; s < M_; ++s) {
        const float lx = sx[last], ly = sy[last], lz = sz[last];
        float bestV = -1.0f;
        int   bestI = 0;
#pragma unroll
        for (int i = 0; i < 4; ++i) {
            const int p = tid + i * 1024;
            const float dx = sx[p] - lx, dy = sy[p] - ly, dz = sz[p] - lz;
            const float d = dx * dx + dy * dy + dz * dz;
            mind[i] = fminf(mind[i], d);
            if (mind[i] > bestV) { bestV = mind[i]; bestI = p; }
        }
        // wave32 argmax reduce (ties -> lowest index, like jnp.argmax)
#pragma unroll
        for (int off = 16; off > 0; off >>= 1) {
            const float ov = __shfl_xor(bestV, off);
            const int   oi = __shfl_xor(bestI, off);
            if (ov > bestV || (ov == bestV && oi < bestI)) { bestV = ov; bestI = oi; }
        }
        if (lane == 0) { redV[wid] = bestV; redI[wid] = bestI; }
        __syncthreads();
        if (tid < 32) {
            float v = redV[tid];
            int   ix = redI[tid];
#pragma unroll
            for (int off = 16; off > 0; off >>= 1) {
                const float ov = __shfl_xor(v, off);
                const int   oi = __shfl_xor(ix, off);
                if (ov > v || (ov == v && oi < ix)) { v = ov; ix = oi; }
            }
            if (tid == 0) { sel = ix; fpsIdx[s] = ix; }
        }
        __syncthreads();
        last = sel;
    }
    __syncthreads();

    for (int s = tid; s < M_; s += 1024) {
        const int i = fpsIdx[s];
        const float x = sx[i], y = sy[i], z = sz[i];
        const size_t wo = ((size_t)(t * B_ + b) * M_ + s) * 3;
        newxyz[wo + 0] = x; newxyz[wo + 1] = y; newxyz[wo + 2] = z;
        const size_t oo = ((size_t)(b * T_ + t) * M_ + s) * 3;
        out_newxyz[oo + 0] = x; out_newxyz[oo + 1] = y; out_newxyz[oo + 2] = z;
    }
}

// ---------------------------------------------------------------------------
// 2) Convert weights to f16 once per launch.
// ---------------------------------------------------------------------------
__global__ __launch_bounds__(256) void wconvert_kernel(const float* __restrict__ w1,
                                                       const float* __restrict__ w2,
                                                       _Float16* __restrict__ w1h,
                                                       _Float16* __restrict__ w2h)
{
    const int i = blockIdx.x * 256 + threadIdx.x;
    if (i < C1 * C0) w1h[i] = (_Float16)w1[i];
    if (i < C2 * C1) w2h[i] = (_Float16)w2[i];
}

__global__ __launch_bounds__(256) void zero_stats_kernel(float* __restrict__ s)
{
    for (int i = threadIdx.x; i < 1536; i += 256) s[i] = 0.0f;
}

// ---------------------------------------------------------------------------
// 3) Ball query + first MLP layer (4->64, no activation), one wave per center.
//    Emits h0 rows (RPF x 64) in f16.
// ---------------------------------------------------------------------------
__global__ __launch_bounds__(256) void ball_h0_kernel(const float* __restrict__ xyzs,
                                                      const float* __restrict__ newxyz,
                                                      const float* __restrict__ w_d,
                                                      _Float16* __restrict__ h0,
                                                      int t)
{
    __shared__ float wd[C0 * 4];
    const int tid = threadIdx.x;
    wd[tid] = w_d[tid];             // 256 == C0*4
    __syncthreads();

    const int lane = tid & 31;
    const int gw = blockIdx.x * 8 + (tid >> 5);
    const int b = gw >> 10;
    const int m = gw & (M_ - 1);

    const float* xyz = xyzs + ((size_t)b * T_ + t) * N_ * 3;
    const size_t co = ((size_t)(t * B_ + b) * M_ + m) * 3;
    const float cx = newxyz[co + 0], cy = newxyz[co + 1], cz = newxyz[co + 2];

    int myN = -1;
    int cnt = 0;
    for (int base = 0; base < N_ && cnt < NS_; base += 32) {
        __builtin_prefetch(xyz + (base + 64) * 3, 0, 0);
        const int n = base + lane;
        const float dx = xyz[n * 3 + 0] - cx;
        const float dy = xyz[n * 3 + 1] - cy;
        const float dz = xyz[n * 3 + 2] - cz;
        const bool in = (dx * dx + dy * dy + dz * dz) < R2_;
        unsigned mask = (unsigned)__builtin_amdgcn_ballot_w32(in);
        while (mask && cnt < NS_) {
            const int src = __builtin_ctz(mask);
            mask &= mask - 1;
            const int nv = base + src;
            if (lane == cnt) myN = nv;
            ++cnt;
        }
    }
    const int first = __shfl(myN, 0);
    if (myN < 0) myN = (first < 0) ? 0 : first;

    const float dx = xyz[myN * 3 + 0] - cx;
    const float dy = xyz[myN * 3 + 1] - cy;
    const float dz = xyz[myN * 3 + 2] - cz;

    const size_t r = ((size_t)(b * M_ + m)) * NS_ + lane;
    _Float16* out = h0 + r * C0;
#pragma unroll
    for (int o8 = 0; o8 < C0; o8 += 8) {
        half8 v;
#pragma unroll
        for (int j = 0; j < 8; ++j) {
            const float* w = &wd[(o8 + j) * 4];
            v[j] = (_Float16)(dx * w[0] + dy * w[1] + dz * w[2]);
        }
        *(half8*)(out + o8) = v;
    }
}

// ---------------------------------------------------------------------------
// WMMA helpers: fragment loads per ISA 7.12.2 layouts (wave32).
// ---------------------------------------------------------------------------
__device__ __forceinline__ v16h make_v16(half8 lo, half8 hi)
{
    v16h r;
#pragma unroll
    for (int e = 0; e < 8; ++e) { r[e] = lo[e]; r[8 + e] = hi[e]; }
    return r;
}

// ---------------------------------------------------------------------------
// 4) GEMM1: h1_pre[r,o] = sum_i h0[r,i] * w1[o,i]  (64 -> 128), f16 WMMA.
//    A tile staged with async global->LDS (ASYNCcnt) — no VGPR round-trip.
//    Accumulates per-channel sum / sumsq for BN1.
// ---------------------------------------------------------------------------
__global__ __launch_bounds__(256) void gemm1_kernel(const _Float16* __restrict__ A,
                                                    const _Float16* __restrict__ W,
                                                    _Float16* __restrict__ Hout,
                                                    float* __restrict__ sumv,
                                                    float* __restrict__ sqv)
{
    __shared__ _Float16 Alds[128 * C0];
    __shared__ float sSum[C1], sSq[C1];
    const int tid = threadIdx.x;
    if (tid < C1) { sSum[tid] = 0.0f; sSq[tid] = 0.0f; }

    const size_t rowBase = (size_t)blockIdx.x * 128;
    const _Float16* Ag = A + rowBase * C0;
    const unsigned ldsBase = (unsigned)(size_t)(&Alds[0]);
#pragma unroll
    for (int i = tid * 8; i < 128 * C0; i += 256 * 8)
        async_copy_b128(ldsBase + (unsigned)(i * 2), Ag + i);
    async_wait0();
    __syncthreads();

    const int wid = tid >> 5, lane = tid & 31;
    const int row0 = wid * 16;
    const int mrow = lane & 15;
    const int hi   = (lane >> 4) * 8;     // A K-half select
    const int ncol = lane & 15;
    const int koff = (lane >> 4) * 8;     // B K-half select
    const int jrow = (lane >> 4) * 8;     // C/D row offset

    const _Float16* arow = Alds + (row0 + mrow) * C0;
    const v16h a0 = make_v16(*(const half8*)(arow + hi),      *(const half8*)(arow + 16 + hi));
    const v16h a1 = make_v16(*(const half8*)(arow + 32 + hi), *(const half8*)(arow + 48 + hi));

    for (int n0 = 0; n0 < C1; n0 += 16) {
        const int ch = n0 + ncol;
        const _Float16* wr = W + (size_t)ch * C0 + koff;
        const v16h bf0 = make_v16(*(const half8*)(wr),      *(const half8*)(wr + 16));
        const v16h bf1 = make_v16(*(const half8*)(wr + 32), *(const half8*)(wr + 48));
        v8f c = {0.f, 0.f, 0.f, 0.f, 0.f, 0.f, 0.f, 0.f};
        c = __builtin_amdgcn_wmma_f32_16x16x32_f16(false, a0, false, bf0, (short)0, c, false, false);
        c = __builtin_amdgcn_wmma_f32_16x16x32_f16(false, a1, false, bf1, (short)0, c, false, false);

        float s = 0.0f, q = 0.0f;
#pragma unroll
        for (int j = 0; j < 8; ++j) {
            const float v = c[j];
            s += v; q += v * v;
            Hout[(rowBase + row0 + jrow + j) * C1 + ch] = (_Float16)v;
        }
        s += __shfl_xor(s, 16);
        q += __shfl_xor(q, 16);
        if (lane < 16) { atomicAdd(&sSum[ch], s); atomicAdd(&sSq[ch], q); }
    }
    __syncthreads();
    if (tid < C1) { atomicAdd(&sumv[tid], sSum[tid]); atomicAdd(&sqv[tid], sSq[tid]); }
}

// ---------------------------------------------------------------------------
// 5) BN finalize: scale/shift from sums.
// ---------------------------------------------------------------------------
__global__ __launch_bounds__(256) void bn_finalize_kernel(const float* __restrict__ sum,
                                                          const float* __restrict__ sq,
                                                          const float* __restrict__ g,
                                                          const float* __restrict__ bb,
                                                          float* __restrict__ sc,
                                                          float* __restrict__ sh,
                                                          int C)
{
    const int c = blockIdx.x * 256 + threadIdx.x;
    if (c >= C) return;
    const float inv = 1.0f / (float)RPF;
    const float mu  = sum[c] * inv;
    const float var = sq[c] * inv - mu * mu;
    const float s   = g[c] * rsqrtf(var + 1e-5f);
    sc[c] = s;
    sh[c] = bb[c] - mu * s;
}

// ---------------------------------------------------------------------------
// 6) GEMM2: A tile async-copied raw to LDS, then BN1+ReLU applied in place;
//    128 -> 256 WMMA. Accumulates per-channel sum / sumsq for BN2.
// ---------------------------------------------------------------------------
__global__ __launch_bounds__(256) void gemm2_kernel(const _Float16* __restrict__ A,
                                                    const _Float16* __restrict__ W,
                                                    _Float16* __restrict__ Hout,
                                                    const float* __restrict__ sc1,
                                                    const float* __restrict__ sh1,
                                                    float* __restrict__ sumv,
                                                    float* __restrict__ sqv)
{
    __shared__ _Float16 Alds[128 * C1];   // 32KB
    __shared__ float sSum[C2], sSq[C2];
    __shared__ float scL[C1], shL[C1];
    const int tid = threadIdx.x;
    if (tid < C1) { scL[tid] = sc1[tid]; shL[tid] = sh1[tid]; }
    sSum[tid] = 0.0f; sSq[tid] = 0.0f;

    const size_t rowBase = (size_t)blockIdx.x * 128;
    const _Float16* Ag = A + rowBase * C1;
    const unsigned ldsBase = (unsigned)(size_t)(&Alds[0]);
#pragma unroll
    for (int i = tid * 8; i < 128 * C1; i += 256 * 8)
        async_copy_b128(ldsBase + (unsigned)(i * 2), Ag + i);
    async_wait0();
    __syncthreads();

    // In-place BN1 + ReLU on the staged tile (each thread rewrites its chunk).
    for (int i = tid * 8; i < 128 * C1; i += 256 * 8) {
        const half8 v = *(const half8*)(Alds + i);
        const int cc = i & (C1 - 1);
        half8 o;
#pragma unroll
        for (int j = 0; j < 8; ++j) {
            float x = (float)v[j];
            x = x * scL[cc + j] + shL[cc + j];
            o[j] = (_Float16)fmaxf(x, 0.0f);
        }
        *(half8*)(Alds + i) = o;
    }
    __syncthreads();

    const int wid = tid >> 5, lane = tid & 31;
    const int row0 = wid * 16;
    const int mrow = lane & 15;
    const int hi   = (lane >> 4) * 8;
    const int ncol = lane & 15;
    const int koff = (lane >> 4) * 8;
    const int jrow = (lane >> 4) * 8;

    const _Float16* arow = Alds + (row0 + mrow) * C1;
    v16h a[4];
#pragma unroll
    for (int kb = 0; kb < 4; ++kb)
        a[kb] = make_v16(*(const half8*)(arow + kb * 32 + hi),
                         *(const half8*)(arow + kb * 32 + 16 + hi));

    for (int n0 = 0; n0 < C2; n0 += 16) {
        const int ch = n0 + ncol;
        const _Float16* wr = W + (size_t)ch * C1 + koff;
        v8f c = {0.f, 0.f, 0.f, 0.f, 0.f, 0.f, 0.f, 0.f};
#pragma unroll
        for (int kb = 0; kb < 4; ++kb) {
            const v16h bf = make_v16(*(const half8*)(wr + kb * 32),
                                     *(const half8*)(wr + kb * 32 + 16));
            c = __builtin_amdgcn_wmma_f32_16x16x32_f16(false, a[kb], false, bf, (short)0, c, false, false);
        }
        float s = 0.0f, q = 0.0f;
#pragma unroll
        for (int j = 0; j < 8; ++j) {
            const float v = c[j];
            s += v; q += v * v;
            Hout[(rowBase + row0 + jrow + j) * C2 + ch] = (_Float16)v;
        }
        s += __shfl_xor(s, 16);
        q += __shfl_xor(q, 16);
        if (lane < 16) { atomicAdd(&sSum[ch], s); atomicAdd(&sSq[ch], q); }
    }
    __syncthreads();
    atomicAdd(&sumv[tid], sSum[tid]);
    atomicAdd(&sqv[tid], sSq[tid]);
}

// ---------------------------------------------------------------------------
// 7) BN2 + ReLU + max over K + transposed store to d_out feat (B,T,256,M).
// ---------------------------------------------------------------------------
__global__ __launch_bounds__(256) void maxpool_kernel(const _Float16* __restrict__ h2,
                                                      const float* __restrict__ sc2,
                                                      const float* __restrict__ sh2,
                                                      float* __restrict__ out_feat,
                                                      int t)
{
    const int bm = blockIdx.x;          // b*M + m
    const int c  = threadIdx.x;
    const int b  = bm >> 10;
    const int m  = bm & (M_ - 1);
    const _Float16* p = h2 + ((size_t)bm * NS_) * C2 + c;
    const float s = sc2[c], h = sh2[c];
    float mv = -1e30f;
#pragma unroll
    for (int k = 0; k < NS_; ++k) {
        const float x = fmaxf((float)p[(size_t)k * C2] * s + h, 0.0f);
        mv = fmaxf(mv, x);
    }
    out_feat[((size_t)(b * T_ + t) * C2 + c) * M_ + m] = mv;
}

// ---------------------------------------------------------------------------
// Host orchestration
// ---------------------------------------------------------------------------
extern "C" void kernel_launch(void* const* d_in, const int* in_sizes, int n_in,
                              void* d_out, int out_size, void* d_ws, size_t ws_size,
                              hipStream_t stream)
{
    (void)in_sizes; (void)n_in; (void)out_size; (void)ws_size;
    const float* xyzs = (const float*)d_in[0];
    const float* w_d  = (const float*)d_in[1];
    const float* w1   = (const float*)d_in[2];
    const float* g1   = (const float*)d_in[3];
    const float* b1   = (const float*)d_in[4];
    const float* w2   = (const float*)d_in[5];
    const float* g2   = (const float*)d_in[6];
    const float* b2   = (const float*)d_in[7];
    float* out = (float*)d_out;
    char*  ws  = (char*)d_ws;

    size_t off = 0;
    float* newxyz = (float*)(ws + off);        off += (size_t)T_ * B_ * M_ * 3 * 4;   // 294912
    float* stats  = (float*)(ws + off);        off += 1536 * 4 + 256;                 // sums+sc/sh
    _Float16* w1h = (_Float16*)(ws + off);     off += (size_t)C1 * C0 * 2;
    _Float16* w2h = (_Float16*)(ws + off);     off += (size_t)C2 * C1 * 2;
    off = (off + 255) & ~(size_t)255;
    _Float16* h0  = (_Float16*)(ws + off);     off += RPF * C0 * 2;
    _Float16* h1  = (_Float16*)(ws + off);     off += RPF * C1 * 2;
    _Float16* h2  = (_Float16*)(ws + off);     off += RPF * C2 * 2;

    // stats layout (floats): sum1[0], sq1[128], sum2[256], sq2[512],
    //                        sc1[768], sh1[896], sc2[1024], sh2[1280]
    float* sum1 = stats + 0,    *sq1 = stats + 128;
    float* sum2 = stats + 256,  *sq2 = stats + 512;
    float* sc1  = stats + 768,  *sh1 = stats + 896;
    float* sc2  = stats + 1024, *sh2 = stats + 1280;

    fps_kernel<<<T_ * B_, 1024, 0, stream>>>(xyzs, newxyz, out);
    wconvert_kernel<<<(C2 * C1 + 255) / 256, 256, 0, stream>>>(w1, w2, w1h, w2h);

    const int gemmBlocks = (int)(RPF / 128);   // 2048
    for (int t = 0; t < T_; ++t) {
        zero_stats_kernel<<<1, 256, 0, stream>>>(stats);
        ball_h0_kernel<<<B_ * M_ / 8, 256, 0, stream>>>(xyzs, newxyz, w_d, h0, t);
        gemm1_kernel<<<gemmBlocks, 256, 0, stream>>>(h0, w1h, h1, sum1, sq1);
        bn_finalize_kernel<<<1, 256, 0, stream>>>(sum1, sq1, g1, b1, sc1, sh1, C1);
        gemm2_kernel<<<gemmBlocks, 256, 0, stream>>>(h1, w2h, h2, sc1, sh1, sum2, sq2);
        bn_finalize_kernel<<<1, 256, 0, stream>>>(sum2, sq2, g2, b2, sc2, sh2, C2);
        maxpool_kernel<<<B_ * M_, 256, 0, stream>>>(h2, sc2, sh2,
                                                    out + (size_t)B_ * T_ * M_ * 3, t);
    }
}
